// DeepRSTPredictor_41566693490828
// MI455X (gfx1250) — matrix operand order repaired and checked by reading, not compiled, not run on hardware
//
#include <hip/hip_runtime.h>
#include <cmath>
#include <complex>
#include <cstring>

// ---------------------------------------------------------------------------
// Types for gfx1250 WMMA (wave32): A/B = 16 halves (8 VGPRs), C/D = 8 floats.
// ---------------------------------------------------------------------------
typedef __attribute__((ext_vector_type(16))) _Float16 v16h;
typedef __attribute__((ext_vector_type(8)))  _Float16 v8h;
typedef __attribute__((ext_vector_type(8)))  float    v8f;

// Explicit LDS (addrspace 3) pointer types: plain ds_load/ds_store, no
// generic-pointer null-check cndmask chains.
typedef __attribute__((address_space(3))) _Float16*       lptr;
typedef const __attribute__((address_space(3))) _Float16* clptr;
typedef const __attribute__((address_space(3))) v8h*      clv8p;
typedef __attribute__((address_space(3))) float*          lfp;

// Explicit GLOBAL (addrspace 1) pointer types: global_load_* instead of
// flat_load_* (flat ops tie up BOTH LOADcnt and DScnt on CDNA5, coupling the
// weight stream to the LDS activation stream; global ops use LOADcnt only).
typedef const __attribute__((address_space(1))) _Float16* gwptr;
typedef const __attribute__((address_space(1))) v8h*      gv8p;
typedef const __attribute__((address_space(1))) float*    gfptr;
typedef __attribute__((address_space(1))) float*          gfout;

// ---------------------------------------------------------------------------
// Converted-weight layout inside d_ws (f16 elements).  All linear weights are
// stored TRANSPOSED as [out][K] so B fragments are contiguous 128-bit loads.
// tp weights stay row-major (u,v): B(k=v,col=u) = w[u*64+v] is contiguous in k.
// ---------------------------------------------------------------------------
constexpr int OFF_L1W0 = 0;                          // 128 x 16
constexpr int OFF_L1W1 = OFF_L1W0 + 16 * 128;        // 256 x 8
constexpr int OFF_L1W2 = OFF_L1W1 + 8 * 256;         // 128 x 8
constexpr int OFF_L2W0 = OFF_L1W2 + 8 * 128;         // 64 x 128
constexpr int OFF_L2W1 = OFF_L2W0 + 128 * 64;        // 128 x 256
constexpr int OFF_L2W2 = OFF_L2W1 + 256 * 128;       // 64 x 128
constexpr int OFF_LW0  = OFF_L2W2 + 128 * 64;        // 11 x (64 x 64)
constexpr int OFF_LW1  = OFF_LW0 + 11 * 64 * 64;     // 11 x (128 x 128)
constexpr int OFF_LW2  = OFF_LW1 + 11 * 128 * 128;   // 11 x (64 x 64)
constexpr int OFF_TPW  = OFF_LW2 + 11 * 64 * 64;     // 3 x 14 x (64 x 64) row-major
constexpr int OFF_TP4  = OFF_TPW + 3 * 14 * 64 * 64; // 17 x (64 x 64) row-major
constexpr int W_TOTAL  = OFF_TP4 + 17 * 64 * 64;     // 566272 f16 elems
constexpr size_t CG_BYTE_OFF = (size_t)W_TOTAL * 2;  // CG tables (729 floats) follow

// LDS carve: hbuf/abuf/bbuf/tbuf/pbuf each 12288 f16, + 96 f32 output accum.
constexpr int SMEM_BYTES = 5 * 12288 * 2 + 96 * 4;   // 123264 B (< 320 KB/WGP)

// ---------------------------------------------------------------------------
// CG table offsets: dense table for all 27 (l1,l2,l3) triples, l in [0,2].
// ---------------------------------------------------------------------------
__host__ __device__ inline int cg_off(int a, int b, int c) {
  int idx = a * 9 + b * 3 + c, off = 0;
  for (int t = 0; t < idx; ++t) {
    int x = t / 9, y = (t / 3) % 3, z = t % 3;
    off += (2 * x + 1) * (2 * y + 1) * (2 * z + 1);
  }
  return off;
}

// ---------------------------------------------------------------------------
// Irrep/path constant tables (HID0 = 64x0 + 64x1 + 64x2 + 64x1)
// ---------------------------------------------------------------------------
__constant__ int c_ls[4]     = {0, 1, 2, 1};
__constant__ int c_cbase[4]  = {0, 1, 4, 9};   // component base index per block
// uvu paths grouped by output block io (weight index p follows UVU_INS order)
__constant__ int c_gnp[4]    = {3, 4, 4, 3};
__constant__ int c_gstart[4] = {0, 3, 7, 11};
__constant__ int c_gp[14]    = {0, 5, 10,   1, 4, 6, 11,   2, 8, 9, 13,   3, 7, 12};
__constant__ int c_gi1[14]   = {0, 1, 2,    0, 1, 1, 2,    0, 2, 2, 2,    0, 1, 2};
__constant__ int c_gi2[14]   = {0, 1, 2,    1, 0, 1, 2,    2, 0, 1, 3,    3, 1, 2};
__constant__ float c_uvu_sc[4] = {0.07216878f, 0.0625f, 0.0625f, 0.07216878f}; // 1/sqrt(fan)
// fully-connected TP paths, reference order (6 paths -> l=0, 11 paths -> l=2)
__constant__ int c_fi1[17] = {0, 1, 1, 2, 3, 3,   0, 1, 1, 1, 2, 2, 2, 2, 3, 3, 3};
__constant__ int c_fi2[17] = {0, 1, 3, 2, 1, 3,   2, 1, 2, 3, 0, 1, 2, 3, 1, 2, 3};

// ---------------------------------------------------------------------------
// One 16x16 D tile:  D = sum_k A(16xK) * B(Kx16), K padded to multiples of 32.
// A from LDS, comp-major (row n, col k), optionally concatenated (A0|A1) on K.
// B from global f16 weights stored transposed: Wt[(col)*w_ld + k].
// ---------------------------------------------------------------------------
__device__ inline v8f mm_tile(int lane,
                              clptr A0, int ka0,
                              clptr A1, int ka1,
                              int a_ld,
                              gwptr Wt, int w_ld,
                              int col0)
{
  v8f acc = {};
  const int K   = ka0 + ka1;
  const int row = lane & 15;
  const int hi  = (lane >> 4) & 1;
  gwptr wrow = Wt + (size_t)(col0 + row) * w_ld;
  for (int k0 = 0; k0 < K; k0 += 32) {
    // concat boundaries are multiples of 32, so a chunk never straddles A0/A1
    clptr Ab = (k0 < ka0) ? (A0 + row * a_ld + k0)
                          : (A1 + row * a_ld + (k0 - ka0));
    v16h a, b;
    if (k0 + 32 <= K) {
      // full chunk: 128-bit LDS loads (A) and 128-bit global loads (B)
      v8h alo = *(clv8p)(Ab + hi * 8);
      v8h ahi = *(clv8p)(Ab + 16 + hi * 8);
      v8h blo = *(gv8p)(wrow + k0 + hi * 8);
      v8h bhi = *(gv8p)(wrow + k0 + 16 + hi * 8);
#pragma unroll
      for (int j = 0; j < 8; ++j) {
        a[j] = alo[j]; a[8 + j] = ahi[j];
        b[j] = blo[j]; b[8 + j] = bhi[j];
      }
    } else {
      const int rem = K - k0; // only happens for single small chunk (K=8/16)
#pragma unroll
      for (int j = 0; j < 16; ++j) {
        int kka = (j < 8) ? (hi * 8 + j) : (8 + hi * 8 + j);
        a[j] = (kka < rem) ? Ab[kka] : (_Float16)0.f;
        int kkb = kka;
        b[j] = (kkb < rem) ? wrow[k0 + kkb] : (_Float16)0.f;
      }
    }
    acc = __builtin_amdgcn_wmma_f32_16x16x32_f16(false, a, false, b,
                                                 (short)0, acc, false, false);
  }
  return acc;
}

// Store D (f32 accum) as f16 to LDS; layout: VGPR r -> row hi*8+r, col lane&15.
// HASB is compile-time so the bias path is branch-free (plain global loads).
template <bool HASB>
__device__ inline void store_tile(lptr dst, int dld, int dcol0, v8f acc,
                                  float scale, gfptr bias, int lane)
{
  const int col = lane & 15, hi = (lane >> 4) & 1;
  float badd = 0.f;
  if constexpr (HASB) badd = bias[col];
#pragma unroll
  for (int r = 0; r < 8; ++r)
    dst[(hi * 8 + r) * dld + dcol0 + col] = (_Float16)(acc[r] * scale + badd);
}

// ---------------------------------------------------------------------------
// Linears.  Activation layouts (f16, comp-major):
//   HID0 buffers: 12 comps of 16x64 (ld 64), comp c at +c*1024
//   HID1 buffer : 12 comps of 16x128 (ld 128), comp c at +c*2048
// ---------------------------------------------------------------------------
__device__ void linear_h0(int lane, clptr src, lptr dst,
                          gwptr W0, gwptr W1, gwptr W2, gfptr bias)
{
  const float s64 = 0.125f, s128 = 0.08838834764f;
  for (int nt = 0; nt < 4; ++nt) {                            // l=0, K=64
    v8f acc = mm_tile(lane, src, 64, nullptr, 0, 64, W0, 64, nt * 16);
    store_tile<true>(dst, 64, nt * 16, acc, s64, bias + nt * 16, lane);
  }
  for (int c = 0; c < 3; ++c)                                 // l=1, K=128 concat
    for (int nt = 0; nt < 8; ++nt) {
      v8f acc = mm_tile(lane, src + (1 + c) * 1024, 64,
                        src + (9 + c) * 1024, 64, 64, W1, 128, nt * 16);
      lptr d = dst + ((nt < 4) ? (1 + c) : (9 + c)) * 1024;
      store_tile<false>(d, 64, (nt & 3) * 16, acc, s128, nullptr, lane);
    }
  for (int c = 0; c < 5; ++c)                                 // l=2, K=64
    for (int nt = 0; nt < 4; ++nt) {
      v8f acc = mm_tile(lane, src + (4 + c) * 1024, 64, nullptr, 0, 64, W2, 64, nt * 16);
      store_tile<false>(dst + (4 + c) * 1024, 64, nt * 16, acc, s64, nullptr, lane);
    }
}

__device__ void linear_in(int lane, clptr px, lptr h1,
                          gwptr W0, gwptr W1, gwptr W2, gfptr b0)
{
  const float s16 = 0.25f, s8 = 0.35355339059f;
  for (int nt = 0; nt < 8; ++nt) {                            // l=0: 16 -> 128
    v8f acc = mm_tile(lane, px, 16, nullptr, 0, 16, W0, 16, nt * 16);
    store_tile<true>(h1, 128, nt * 16, acc, s16, b0 + nt * 16, lane);
  }
  for (int c = 0; c < 3; ++c)                                 // l=1: 8 -> 128+128
    for (int nt = 0; nt < 16; ++nt) {
      v8f acc = mm_tile(lane, px + 256 + c * 128, 8, nullptr, 0, 8, W1, 8, nt * 16);
      lptr d = h1 + ((nt < 8) ? (1 + c) : (9 + c)) * 2048;
      store_tile<false>(d, 128, (nt & 7) * 16, acc, s8, nullptr, lane);
    }
  for (int c = 0; c < 5; ++c)                                 // l=2: 8 -> 128
    for (int nt = 0; nt < 8; ++nt) {
      v8f acc = mm_tile(lane, px + 640 + c * 128, 8, nullptr, 0, 8, W2, 8, nt * 16);
      store_tile<false>(h1 + (4 + c) * 2048, 128, nt * 16, acc, s8, nullptr, lane);
    }
}

__device__ void linear_h1h0(int lane, clptr h1, lptr dst,
                            gwptr W0, gwptr W1, gwptr W2, gfptr b0)
{
  const float s128 = 0.08838834764f, s256 = 0.0625f;
  for (int nt = 0; nt < 4; ++nt) {                            // l=0: 128 -> 64
    v8f acc = mm_tile(lane, h1, 128, nullptr, 0, 128, W0, 128, nt * 16);
    store_tile<true>(dst, 64, nt * 16, acc, s128, b0 + nt * 16, lane);
  }
  for (int c = 0; c < 3; ++c)                                 // l=1: 256 -> 128
    for (int nt = 0; nt < 8; ++nt) {
      v8f acc = mm_tile(lane, h1 + (1 + c) * 2048, 128,
                        h1 + (9 + c) * 2048, 128, 128, W1, 256, nt * 16);
      lptr d = dst + ((nt < 4) ? (1 + c) : (9 + c)) * 1024;
      store_tile<false>(d, 64, (nt & 3) * 16, acc, s256, nullptr, lane);
    }
  for (int c = 0; c < 5; ++c)                                 // l=2: 128 -> 64
    for (int nt = 0; nt < 4; ++nt) {
      v8f acc = mm_tile(lane, h1 + (4 + c) * 2048, 128, nullptr, 0, 128, W2, 128, nt * 16);
      store_tile<false>(dst + (4 + c) * 1024, 64, nt * 16, acc, s128, nullptr, lane);
    }
}

// ---------------------------------------------------------------------------
// uvu tensor product: per output block io, GEMM all contributing paths'
// b_p = w_p . x2 (WMMA) into pbuf, then CG-contract on VALU with f32 regs.
// ---------------------------------------------------------------------------
__device__ void tp_uvu_dev(int lane, clptr x1, clptr x2,
                           gwptr wp, lptr dst, lptr pb, gfptr cgt)
{
  for (int io = 0; io < 4; ++io) {
    const int lo = c_ls[io], nko = 2 * lo + 1;
    const int np = c_gnp[io], g0 = c_gstart[io];
    int slot = 0;
    for (int pp = 0; pp < np; ++pp) {                 // GEMMs: b_p[n,u,j]
      const int gi = g0 + pp;
      const int i2 = c_gi2[gi], nj = 2 * c_ls[i2] + 1;
      gwptr w = wp + c_gp[gi] * 4096;                 // (u,v) row-major == Wt
      for (int j = 0; j < nj; ++j) {
        clptr A = x2 + (c_cbase[i2] + j) * 1024;
        for (int nt = 0; nt < 4; ++nt) {
          v8f acc = mm_tile(lane, A, 64, nullptr, 0, 64, w, 64, nt * 16);
          store_tile<false>(pb + (slot + j) * 1024, 64, nt * 16, acc, 1.f, nullptr, lane);
        }
      }
      slot += nj;
    }
    const float sc = c_uvu_sc[io];
    for (int t = 0; t < 32; ++t) {                    // CG contraction per (n,u)
      const int q = lane + 32 * t;                    // q = n*64 + u
      float acc[5];
#pragma unroll
      for (int k = 0; k < 5; ++k) acc[k] = 0.f;
      int slot2 = 0;
      for (int pp = 0; pp < np; ++pp) {
        const int gi = g0 + pp;
        const int i1 = c_gi1[gi], i2 = c_gi2[gi];
        const int l1 = c_ls[i1], l2 = c_ls[i2];
        const int ni = 2 * l1 + 1, nj = 2 * l2 + 1;
        gfptr Kt = cgt + cg_off(l1, l2, lo);
        float xv[5], bv[5];
#pragma unroll
        for (int i = 0; i < 5; ++i)
          xv[i] = (i < ni) ? (float)x1[(c_cbase[i1] + i) * 1024 + q] : 0.f;
#pragma unroll
        for (int j = 0; j < 5; ++j)
          bv[j] = (j < nj) ? (float)pb[(slot2 + j) * 1024 + q] : 0.f;
#pragma unroll
        for (int i = 0; i < 5; ++i)
#pragma unroll
          for (int j = 0; j < 5; ++j)
            if (i < ni && j < nj) {
              const float xb = xv[i] * bv[j];
#pragma unroll
              for (int k = 0; k < 5; ++k)
                if (k < nko) acc[k] += xb * Kt[(i * nj + j) * nko + k];
            }
        slot2 += nj;
      }
#pragma unroll
      for (int k = 0; k < 5; ++k)
        if (k < nko) dst[(c_cbase[io] + k) * 1024 + q] = (_Float16)(acc[k] * sc);
    }
  }
}

// Fully-connected TP -> 6 outputs per node (sum over u via ds_add_f32 atomics).
__device__ void tp_fc_dev(int lane, clptr x1, clptr x2,
                          gwptr wp, lptr pb, lfp oacc, gfptr cgt,
                          gfout outg, int n0)
{
  for (int idx = lane; idx < 96; idx += 32) oacc[idx] = 0.f;
  for (int p = 0; p < 17; ++p) {
    const int i1 = c_fi1[p], i2 = c_fi2[p];
    const int l1 = c_ls[i1], l2 = c_ls[i2];
    const int lo = (p < 6) ? 0 : 2, nko = 2 * lo + 1, koff = (p < 6) ? 0 : 1;
    const int ni = 2 * l1 + 1, nj = 2 * l2 + 1;
    gwptr w = wp + p * 4096;
    for (int j = 0; j < nj; ++j)
      for (int nt = 0; nt < 4; ++nt) {
        v8f acc = mm_tile(lane, x2 + (c_cbase[i2] + j) * 1024, 64,
                          nullptr, 0, 64, w, 64, nt * 16);
        store_tile<false>(pb + j * 1024, 64, nt * 16, acc, 1.f, nullptr, lane);
      }
    gfptr Kt = cgt + cg_off(l1, l2, lo);
    for (int t = 0; t < 32; ++t) {
      const int q = lane + 32 * t, n = q >> 6;
      float xv[5], bv[5], acc[5];
#pragma unroll
      for (int i = 0; i < 5; ++i)
        xv[i] = (i < ni) ? (float)x1[(c_cbase[i1] + i) * 1024 + q] : 0.f;
#pragma unroll
      for (int j = 0; j < 5; ++j)
        bv[j] = (j < nj) ? (float)pb[j * 1024 + q] : 0.f;
#pragma unroll
      for (int k = 0; k < 5; ++k) acc[k] = 0.f;
#pragma unroll
      for (int i = 0; i < 5; ++i)
#pragma unroll
        for (int j = 0; j < 5; ++j)
          if (i < ni && j < nj) {
            const float xb = xv[i] * bv[j];
#pragma unroll
            for (int k = 0; k < 5; ++k)
              if (k < nko) acc[k] += xb * Kt[(i * nj + j) * nko + k];
          }
#pragma unroll
      for (int k = 0; k < 5; ++k)
        if (k < nko)
          __hip_atomic_fetch_add(&oacc[n * 6 + koff + k], acc[k],
                                 __ATOMIC_RELAXED, __HIP_MEMORY_SCOPE_WORKGROUP);
    }
  }
  const float s0 = 1.f / sqrtf(24576.f);   // fan io=0: 6*64*64
  const float s1 = 1.f / sqrtf(45056.f);   // fan io=1: 11*64*64
  for (int idx = lane; idx < 96; idx += 32) {
    int n = idx / 6, k = idx % 6;
    outg[(size_t)(n0 + n) * 6 + k] = oacc[idx] * ((k == 0) ? s0 : s1);
  }
}

// ---------------------------------------------------------------------------
// Fused kernel: one wave32 per 16-node tile; whole network stays in LDS.
// ---------------------------------------------------------------------------
extern "C" __global__ void __launch_bounds__(32)
e3nn_deeprst_kernel(const float* __restrict__ x_,
                    const float* __restrict__ lin1_b0_,
                    const float* __restrict__ lin2_b0_,
                    const float* __restrict__ L_b0_,
                    const _Float16* __restrict__ W_,
                    const float* __restrict__ cgt_,
                    float* __restrict__ out_)
{
  // One no-op addrspacecast each: generic -> global (same representation).
  gfptr x       = (gfptr)x_;
  gfptr lin1_b0 = (gfptr)lin1_b0_;
  gfptr lin2_b0 = (gfptr)lin2_b0_;
  gfptr L_b0    = (gfptr)L_b0_;
  gwptr W       = (gwptr)W_;
  gfptr cgt     = (gfptr)cgt_;
  gfout out     = (gfout)out_;

  extern __shared__ _Float16 smem[];
  lptr hbuf = (lptr)smem;                 // current h (HID0)
  lptr abuf = hbuf + 12288;               // ha   (also HID1 lo half)
  lptr bbuf = abuf + 12288;               // hb   (also HID1 hi half)
  lptr tbuf = bbuf + 12288;               // tp output
  lptr pbuf = tbuf + 12288;               // path scratch (also x staging)
  lfp  oacc = (lfp)(pbuf + 12288);
  lptr h1 = abuf;                         // 12 comps x 16x128 spans abuf+bbuf
  lptr px = pbuf;

  const int lane = threadIdx.x;
  const int n0 = blockIdx.x * 16;

  // stage x (16 x 80 f32) -> comp-major f16: l0 16x16 @0, l1 16x8 @256+i*128,
  // l2 16x8 @640+i*128
  gfptr xg = x + (size_t)n0 * 80;
  for (int idx = lane; idx < 16 * 80; idx += 32) {
    const int n = idx / 80, c = idx % 80;
    const float v = xg[idx];
    int d;
    if (c < 16)      d = n * 16 + c;
    else if (c < 40) { int u = (c - 16) / 3, i = (c - 16) % 3; d = 256 + i * 128 + n * 8 + u; }
    else             { int u = (c - 40) / 5, i = (c - 40) % 5; d = 640 + i * 128 + n * 8 + u; }
    px[d] = (_Float16)v;
  }

  linear_in(lane, px, h1, W + OFF_L1W0, W + OFF_L1W1, W + OFF_L1W2, lin1_b0);
  linear_h1h0(lane, h1, hbuf, W + OFF_L2W0, W + OFF_L2W1, W + OFF_L2W2, lin2_b0);

  for (int it = 0; it < 3; ++it) {
    const int la = 3 * it, lb = 3 * it + 1, lc = 3 * it + 2;
    linear_h0(lane, hbuf, abuf, W + OFF_LW0 + la * 4096, W + OFF_LW1 + la * 16384,
              W + OFF_LW2 + la * 4096, L_b0 + la * 64);
    linear_h0(lane, hbuf, bbuf, W + OFF_LW0 + lb * 4096, W + OFF_LW1 + lb * 16384,
              W + OFF_LW2 + lb * 4096, L_b0 + lb * 64);
    tp_uvu_dev(lane, abuf, bbuf, W + OFF_TPW + it * 14 * 4096, tbuf, pbuf, cgt);
    linear_h0(lane, tbuf, hbuf, W + OFF_LW0 + lc * 4096, W + OFF_LW1 + lc * 16384,
              W + OFF_LW2 + lc * 4096, L_b0 + lc * 64);
  }
  linear_h0(lane, hbuf, abuf, W + OFF_LW0 + 9 * 4096, W + OFF_LW1 + 9 * 16384,
            W + OFF_LW2 + 9 * 4096, L_b0 + 9 * 64);
  linear_h0(lane, hbuf, bbuf, W + OFF_LW0 + 10 * 4096, W + OFF_LW1 + 10 * 16384,
            W + OFF_LW2 + 10 * 4096, L_b0 + 10 * 64);
  tp_fc_dev(lane, abuf, bbuf, W + OFF_TP4, pbuf, oacc, cgt, out, n0);
}

// ---------------------------------------------------------------------------
// Weight conversion kernels: fp32 -> f16 (transposed / straight)
// ---------------------------------------------------------------------------
extern "C" __global__ void convTB(const float* __restrict__ src,
                                  _Float16* __restrict__ dst, int K, int N)
{
  const size_t base = (size_t)blockIdx.y * K * N;
  const int i = blockIdx.x * 256 + threadIdx.x;
  if (i >= K * N) return;
  const int k = i / N, o = i % N;
  dst[base + (size_t)o * K + k] = (_Float16)src[base + i];
}

extern "C" __global__ void convC(const float* __restrict__ src,
                                 _Float16* __restrict__ dst, int n)
{
  const int i = blockIdx.x * 256 + threadIdx.x;
  if (i < n) dst[i] = (_Float16)src[i];
}

// ---------------------------------------------------------------------------
// Host-side real Clebsch-Gordan tables (exact mirror of the reference)
// ---------------------------------------------------------------------------
static double fd(int n) { double r = 1; for (int i = 2; i <= n; ++i) r *= i; return r; }

static void r2c(int l, std::complex<double> U[5][5]) {
  for (int a = 0; a < 5; ++a) for (int b = 0; b < 5; ++b) U[a][b] = 0.0;
  const std::complex<double> I(0, 1);
  U[l][l] = 1.0;
  const double s = 1.0 / std::sqrt(2.0);
  for (int m = 1; m <= l; ++m) {
    const double pm = (m & 1) ? -1.0 : 1.0;
    U[l - m][l + m] = s;
    U[l + m][l + m] = pm * s;
    U[l - m][l - m] = I * s;
    U[l + m][l - m] = -I * pm * s;
  }
  std::complex<double> ph(1, 0);
  for (int t = 0; t < l; ++t) ph *= std::complex<double>(0, -1);
  for (int a = 0; a <= 2 * l; ++a) for (int b = 0; b <= 2 * l; ++b) U[a][b] *= ph;
}

static void build_cg(float* out729) {
  for (int i = 0; i < 729; ++i) out729[i] = 0.f;
  for (int l1 = 0; l1 <= 2; ++l1)
    for (int l2 = 0; l2 <= 2; ++l2)
      for (int l3 = 0; l3 <= 2; ++l3) {
        if (l3 < std::abs(l1 - l2) || l3 > l1 + l2) continue;
        double C[5][5][5] = {};
        for (int m1 = -l1; m1 <= l1; ++m1)
          for (int m2 = -l2; m2 <= l2; ++m2) {
            const int m3 = m1 + m2;
            if (std::abs(m3) > l3) continue;
            double pref = std::sqrt((2 * l3 + 1) * fd(l3 + l1 - l2) * fd(l3 - l1 + l2) *
                                    fd(l1 + l2 - l3) / fd(l1 + l2 + l3 + 1));
            pref *= std::sqrt(fd(l3 + m3) * fd(l3 - m3) * fd(l1 - m1) * fd(l1 + m1) *
                              fd(l2 - m2) * fd(l2 + m2));
            double s = 0.0;
            for (int k = 0; k <= l1 + l2 - l3; ++k) {
              const int d0 = k, d1 = l1 + l2 - l3 - k, d2 = l1 - m1 - k,
                        d3 = l2 + m2 - k, d4 = l3 - l2 + m1 + k, d5 = l3 - l1 - m2 + k;
              if (d0 < 0 || d1 < 0 || d2 < 0 || d3 < 0 || d4 < 0 || d5 < 0) continue;
              s += ((k & 1) ? -1.0 : 1.0) / (fd(d0) * fd(d1) * fd(d2) * fd(d3) * fd(d4) * fd(d5));
            }
            C[m1 + l1][m2 + l2][m3 + l3] = pref * s;
          }
        std::complex<double> U1[5][5], U2[5][5], U3[5][5];
        r2c(l1, U1); r2c(l2, U2); r2c(l3, U3);
        const int n1 = 2 * l1 + 1, n2 = 2 * l2 + 1, n3 = 2 * l3 + 1;
        double Kr[5][5][5], Ki[5][5][5], nr = 0, ni = 0;
        for (int a = 0; a < n1; ++a)
          for (int b = 0; b < n2; ++b)
            for (int c = 0; c < n3; ++c) {
              std::complex<double> acc(0, 0);
              for (int m = 0; m < n1; ++m)
                for (int n = 0; n < n2; ++n)
                  for (int o = 0; o < n3; ++o)
                    acc += U1[m][a] * U2[n][b] * std::conj(U3[o][c]) * C[m][n][o];
              Kr[a][b][c] = acc.real(); Ki[a][b][c] = acc.imag();
              nr += acc.real() * acc.real(); ni += acc.imag() * acc.imag();
            }
        const bool useR = nr >= ni;
        const double nrm = std::sqrt(useR ? nr : ni);
        const double tgt = std::sqrt((double)n3);
        float* dst = out729 + cg_off(l1, l2, l3);
        for (int a = 0; a < n1; ++a)
          for (int b = 0; b < n2; ++b)
            for (int c = 0; c < n3; ++c)
              dst[(a * n2 + b) * n3 + c] =
                  (float)((useR ? Kr[a][b][c] : Ki[a][b][c]) / nrm * tgt);
      }
}

static float g_cg[729];  // static storage: stays valid for graph-replayed H2D copy

// ---------------------------------------------------------------------------
// Launch
// ---------------------------------------------------------------------------
extern "C" void kernel_launch(void* const* d_in, const int* in_sizes, int n_in,
                              void* d_out, int out_size, void* d_ws, size_t ws_size,
                              hipStream_t stream)
{
  (void)n_in; (void)out_size; (void)ws_size;
  const float* x       = (const float*)d_in[0];
  const float* l1W0    = (const float*)d_in[1];
  const float* l1W1    = (const float*)d_in[2];
  const float* l1W2    = (const float*)d_in[3];
  const float* l1b0    = (const float*)d_in[4];
  const float* l2W0    = (const float*)d_in[5];
  const float* l2W1    = (const float*)d_in[6];
  const float* l2W2    = (const float*)d_in[7];
  const float* l2b0    = (const float*)d_in[8];
  const float* LW0     = (const float*)d_in[9];
  const float* LW1     = (const float*)d_in[10];
  const float* LW2     = (const float*)d_in[11];
  const float* Lb0     = (const float*)d_in[12];
  const float* tpw     = (const float*)d_in[13];
  const float* tp4w    = (const float*)d_in[14];

  _Float16* wsF  = (_Float16*)d_ws;
  float*    cgD  = (float*)((char*)d_ws + CG_BYTE_OFF);

  // CG tables: recomputed deterministically each call, copied to d_ws
  build_cg(g_cg);
  hipMemcpyAsync(cgD, g_cg, sizeof(g_cg), hipMemcpyHostToDevice, stream);

  // fp32 -> f16 weight conversion (+ transpose for all linear weights)
  convTB<<<dim3((16 * 128 + 255) / 256, 1), 256, 0, stream>>>(l1W0, wsF + OFF_L1W0, 16, 128);
  convTB<<<dim3((8 * 256 + 255) / 256, 1), 256, 0, stream>>>(l1W1, wsF + OFF_L1W1, 8, 256);
  convTB<<<dim3((8 * 128 + 255) / 256, 1), 256, 0, stream>>>(l1W2, wsF + OFF_L1W2, 8, 128);
  convTB<<<dim3((128 * 64 + 255) / 256, 1), 256, 0, stream>>>(l2W0, wsF + OFF_L2W0, 128, 64);
  convTB<<<dim3((256 * 128 + 255) / 256, 1), 256, 0, stream>>>(l2W1, wsF + OFF_L2W1, 256, 128);
  convTB<<<dim3((128 * 64 + 255) / 256, 1), 256, 0, stream>>>(l2W2, wsF + OFF_L2W2, 128, 64);
  convTB<<<dim3((64 * 64 + 255) / 256, 11), 256, 0, stream>>>(LW0, wsF + OFF_LW0, 64, 64);
  convTB<<<dim3((128 * 128 + 255) / 256, 11), 256, 0, stream>>>(LW1, wsF + OFF_LW1, 128, 128);
  convTB<<<dim3((64 * 64 + 255) / 256, 11), 256, 0, stream>>>(LW2, wsF + OFF_LW2, 64, 64);
  convC<<<(3 * 14 * 4096 + 255) / 256, 256, 0, stream>>>(tpw, wsF + OFF_TPW, 3 * 14 * 4096);
  convC<<<(17 * 4096 + 255) / 256, 256, 0, stream>>>(tp4w, wsF + OFF_TP4, 17 * 4096);

  // allow >64KB dynamic LDS
  hipFuncSetAttribute(reinterpret_cast<const void*>(&e3nn_deeprst_kernel),
                      hipFuncAttributeMaxDynamicSharedMemorySize, SMEM_BYTES);

  const int nodes = in_sizes[0] / 80;
  const int nblk  = nodes / 16;
  e3nn_deeprst_kernel<<<nblk, 32, SMEM_BYTES, stream>>>(
      x, l1b0, l2b0, Lb0, wsF, cgD, (float*)d_out);
}
